// DynamicMixtureOfTasks_83726092468464
// MI455X (gfx1250) — matrix operand order repaired and checked by reading, not compiled, not run on hardware
//
#include <hip/hip_runtime.h>
#include <hip/hip_bf16.h>

// Problem dims (fixed by the reference)
constexpr int kD = 1024, kH = 128, kC = 256, kT = 16, kB = 8192;

// Padded LDS strides (bank-conflict-free for 16B ds loads)
constexpr int kWP = 40;    // W1 k-chunk row stride (elements): 80B -> lane banks 20i mod 64
constexpr int kHP = 136;   // h tile row stride (elements): 272B -> lane banks 4i mod 64

typedef __attribute__((ext_vector_type(16))) __bf16 v16bf;
typedef __attribute__((ext_vector_type(8)))  __bf16 v8bf;
typedef __attribute__((ext_vector_type(4)))  __bf16 v4bf;
typedef __attribute__((ext_vector_type(8)))  float  v8f;
typedef int v4i __attribute__((vector_size(16)));   // matches async builtin param type

#define AS_G __attribute__((address_space(1)))
#define AS_L __attribute__((address_space(3)))

#if defined(__has_builtin)
#if __has_builtin(__builtin_amdgcn_global_load_async_to_lds_b128) && \
    __has_builtin(__builtin_amdgcn_s_wait_asynccnt)
#define USE_ASYNC_LDS 1
#endif
#endif

static __device__ __forceinline__ v16bf cat16(v8bf lo, v8bf hi) {
    return __builtin_shufflevector(lo, hi, 0,1,2,3,4,5,6,7,8,9,10,11,12,13,14,15);
}

// ---------------------------------------------------------------------------
// 1) x: f32 [B,D] -> bf16 [B,D]
// ---------------------------------------------------------------------------
__global__ void __launch_bounds__(256) cvt_x_kernel(const float* __restrict__ x,
                                                    __bf16* __restrict__ xb) {
    size_t i = ((size_t)blockIdx.x * 256 + threadIdx.x) * 4;
    float4 v = *(const float4*)(x + i);
    v4bf o;
    o[0] = (__bf16)v.x; o[1] = (__bf16)v.y; o[2] = (__bf16)v.z; o[3] = (__bf16)v.w;
    *(v4bf*)(xb + i) = o;
}

// ---------------------------------------------------------------------------
// 2) Tiled transpose+convert: src f32 [T][R][Cc] -> dst bf16 [T][Cc][R]
//    grid = (R/32, Cc/32, T), block = 256
// ---------------------------------------------------------------------------
__global__ void __launch_bounds__(256) transpose_cvt_kernel(const float* __restrict__ src,
                                                            __bf16* __restrict__ dst,
                                                            int R, int Cc) {
    __shared__ __bf16 tile[32][33];
    const int t  = blockIdx.z;
    const int r0 = blockIdx.x * 32;
    const int c0 = blockIdx.y * 32;
    const int cc = threadIdx.x & 31;
    const int rr = threadIdx.x >> 5;           // 0..7
    const float* s = src + ((size_t)t * R + r0) * Cc + c0;
#pragma unroll
    for (int i = 0; i < 4; ++i)
        tile[rr + 8 * i][cc] = (__bf16)s[(size_t)(rr + 8 * i) * Cc + cc];
    __syncthreads();
    __bf16* d = dst + ((size_t)t * Cc + c0) * R + r0;
#pragma unroll
    for (int i = 0; i < 4; ++i)
        d[(size_t)(rr + 8 * i) * R + cc] = tile[cc][rr + 8 * i];
}

// ---------------------------------------------------------------------------
// 3) Router: logits = x@Wr + br, softmax over T.  One block = 16 rows of x.
// ---------------------------------------------------------------------------
__global__ void __launch_bounds__(256) router_kernel(const float* __restrict__ x,
                                                     const float* __restrict__ Wr,
                                                     const float* __restrict__ br,
                                                     float* __restrict__ probs) {
    __shared__ float xs[16 * kD];              // 64 KB x tile
    const int tid = threadIdx.x;
    const int b0  = blockIdx.x * 16;
    for (int i = tid * 4; i < 16 * kD; i += 256 * 4)
        *(float4*)&xs[i] = *(const float4*)&x[(size_t)b0 * kD + i];
    __syncthreads();

    const int bl = tid >> 4;                   // local row 0..15
    const int tt = tid & 15;                   // task 0..15
    const float* xr = xs + bl * kD;
    float acc = br[tt];
#pragma unroll 4
    for (int d = 0; d < kD; d += 4) {
        float4 xv = *(const float4*)&xr[d];
        acc = fmaf(xv.x, Wr[(d + 0) * kT + tt], acc);
        acc = fmaf(xv.y, Wr[(d + 1) * kT + tt], acc);
        acc = fmaf(xv.z, Wr[(d + 2) * kT + tt], acc);
        acc = fmaf(xv.w, Wr[(d + 3) * kT + tt], acc);
    }
    float m = acc;
#pragma unroll
    for (int off = 8; off; off >>= 1) m = fmaxf(m, __shfl_xor(m, off, 16));
    float e = __expf(acc - m);
    float s = e;
#pragma unroll
    for (int off = 8; off; off >>= 1) s += __shfl_xor(s, off, 16);
    probs[(size_t)(b0 + bl) * kT + tt] = e / s;
}

// ---------------------------------------------------------------------------
// 4) Fused expert: per (64-row tile, task): h = relu(x@W1+b1) in LDS (bf16),
//    preds = (h@W2 + b2) * prob, written f32 (non-temporal).  8 waves,
//    WMMA bf16 16x16x32.  W1 k-chunks triple-buffered in LDS via
//    GLOBAL_LOAD_ASYNC_TO_LDS_B128 (ASYNCcnt): chunk i+2's DMA issued while
//    computing chunk i, waited with s_wait_asynccnt 2 (2 async ops per chunk,
//    in-order completion) -> two-iteration latency window.
//    Fragment layouts used here (A and B share one contiguous k order):
//      A (16x32, 16-bit): M = lane&15; K run  [k0+16h, +16)
//      B (32x16, 16-bit): N = lane&15; K run  [k0+16h, +16)
//      C/D (16x16 f32):   M = elem + 8h;  N = lane&15          (h = lane>>4)
// ---------------------------------------------------------------------------
__global__ void __launch_bounds__(256) expert_kernel(const __bf16* __restrict__ xb,   // [B,D]
                                                     const __bf16* __restrict__ w1t,  // [T,H,D]
                                                     const __bf16* __restrict__ w2t,  // [T,C,H]
                                                     const float* __restrict__ b1,    // [T,H]
                                                     const float* __restrict__ b2,    // [T,C]
                                                     const float* __restrict__ probs, // [B,T]
                                                     float* __restrict__ out)         // [T,B,C]
{
    __shared__ __bf16 lds_w[3][kH * kWP];   // W1 k-chunk triple buffer (3 x 10 KB)
    __shared__ __bf16 lds_h[64 * kHP];      // h tile [m][n], padded, ~17 KB
    __shared__ float  lds_p[64];
    __shared__ float  lds_b1[kH];
    __shared__ float  lds_b2[kC];

    const int t   = blockIdx.y;
    const int m0  = blockIdx.x * 64;
    const int tid = threadIdx.x;
    const int wave  = tid >> 5;
    const int lane  = tid & 31;
    const int lh    = lane >> 4;        // half-wave
    const int ln    = lane & 15;
    const int strip = wave >> 1;        // 0..3 -> 16-row strip
    const int chalf = wave & 1;         // column half

    // stage per-tile constants
    if (tid < 64)  lds_p[tid]  = probs[(size_t)(m0 + tid) * kT + t];
    if (tid < kH)  lds_b1[tid] = b1[t * kH + tid];
    lds_b2[tid] = b2[t * kC + tid];     // tid covers 0..255

    // per-thread W1 staging assignment: row sn, 16-element half sp
    const int sn = tid >> 1, sp = tid & 1;
    const __bf16* wsrc = w1t + ((size_t)(t * kH + sn)) * kD + 16 * sp;
    const int ldw_off = sn * kWP + 16 * sp;             // elements into lds_w[buf]

    // ---------------- fc1 ---------------------------------------------------
    v8f acc[4] = {};
    const __bf16* xrow = xb + (size_t)(m0 + strip * 16 + ln) * kD;
    constexpr int NCH = kD / 32;        // 32 k-chunks

#if defined(USE_ASYNC_LDS)
    auto issue_chunk = [&](int buf, int k0) {
        uintptr_t g = (uintptr_t)(wsrc + k0);
        uint32_t  l = (uint32_t)(uintptr_t)(&lds_w[buf][ldw_off]);
        __builtin_amdgcn_global_load_async_to_lds_b128(
            (AS_G v4i*)g,        (AS_L v4i*)l,                   0, 0);
        __builtin_amdgcn_global_load_async_to_lds_b128(
            (AS_G v4i*)(g + 16), (AS_L v4i*)(uint32_t)(l + 16), 0, 0);
    };
    issue_chunk(0, 0);
    issue_chunk(1, 32);
    for (int it = 0; it < NCH; ++it) {
        const int k0 = it * 32;
        if (it + 1 < NCH) __builtin_amdgcn_s_wait_asynccnt(2);  // chunk `it` landed
        else              __builtin_amdgcn_s_wait_asynccnt(0);  // last chunk
        __syncthreads();                                        // visible to all waves
        if (it + 2 < NCH) issue_chunk((it + 2) % 3, k0 + 64);   // overlap DMA 2 deep
        const __bf16* wb = lds_w[it % 3];

        v8bf a_lo = *(const v8bf*)(xrow + k0 + 16 * lh);
        v8bf a_hi = *(const v8bf*)(xrow + k0 + 16 * lh + 8);
        v16bf afrag = cat16(a_lo, a_hi);
#pragma unroll
        for (int j = 0; j < 4; ++j) {
            const int n = chalf * 64 + j * 16 + ln;
            v8bf b_lo = *(const v8bf*)(wb + n * kWP + 16 * lh);
            v8bf b_hi = *(const v8bf*)(wb + n * kWP + 16 * lh + 8);
            v16bf bfrag = cat16(b_lo, b_hi);
            acc[j] = __builtin_amdgcn_wmma_f32_16x16x32_bf16(
                false, afrag, false, bfrag, (short)0, acc[j], false, false);
        }
    }
#else
    for (int k0 = 0; k0 < kD; k0 += 32) {
        __syncthreads();
        {   // cooperative sync stage of W1t[t][0..127][k0..k0+32) -> lds_w[0]
            const uint4* src = (const uint4*)(wsrc + k0);
            uint4* dst = (uint4*)(&lds_w[0][ldw_off]);
            dst[0] = src[0];
            dst[1] = src[1];
        }
        __syncthreads();
        const __bf16* wb = lds_w[0];

        v8bf a_lo = *(const v8bf*)(xrow + k0 + 16 * lh);
        v8bf a_hi = *(const v8bf*)(xrow + k0 + 16 * lh + 8);
        v16bf afrag = cat16(a_lo, a_hi);
#pragma unroll
        for (int j = 0; j < 4; ++j) {
            const int n = chalf * 64 + j * 16 + ln;
            v8bf b_lo = *(const v8bf*)(wb + n * kWP + 16 * lh);
            v8bf b_hi = *(const v8bf*)(wb + n * kWP + 16 * lh + 8);
            v16bf bfrag = cat16(b_lo, b_hi);
            acc[j] = __builtin_amdgcn_wmma_f32_16x16x32_bf16(
                false, afrag, false, bfrag, (short)0, acc[j], false, false);
        }
    }
#endif
    __syncthreads();

    // bias + relu -> bf16 h tile in LDS
#pragma unroll
    for (int j = 0; j < 4; ++j) {
        const int n = chalf * 64 + j * 16 + ln;
        const float bias = lds_b1[n];
#pragma unroll
        for (int e = 0; e < 8; ++e) {
            const int m = strip * 16 + e + 8 * lh;
            float v = acc[j][e] + bias;
            v = v > 0.0f ? v : 0.0f;
            lds_h[m * kHP + n] = (__bf16)v;
        }
    }
    __syncthreads();

    // ---------------- fc2: 16 rows x 128 cols per wave ---------------------
    v8f acc2[8] = {};
    const __bf16* hrow = lds_h + (strip * 16 + ln) * kHP;
#pragma unroll
    for (int k0 = 0; k0 < kH; k0 += 32) {
        v8bf a_lo = *(const v8bf*)(hrow + k0 + 16 * lh);
        v8bf a_hi = *(const v8bf*)(hrow + k0 + 16 * lh + 8);
        v16bf afrag = cat16(a_lo, a_hi);
#pragma unroll
        for (int j = 0; j < 8; ++j) {
            const int c = chalf * 128 + j * 16 + ln;
            const __bf16* wrow = w2t + ((size_t)(t * kC + c)) * kH + k0 + 16 * lh;
            v8bf b_lo = *(const v8bf*)(wrow);
            v8bf b_hi = *(const v8bf*)(wrow + 8);
            v16bf bfrag = cat16(b_lo, b_hi);
            acc2[j] = __builtin_amdgcn_wmma_f32_16x16x32_bf16(
                false, afrag, false, bfrag, (short)0, acc2[j], false, false);
        }
    }

    // epilogue: (acc + b2) * prob -> out[t][m][c]; non-temporal (write-once
    // 134MB stream, keep L2 for the reused bf16 operands)
    float pv[8];
#pragma unroll
    for (int e = 0; e < 8; ++e) pv[e] = lds_p[strip * 16 + e + 8 * lh];
#pragma unroll
    for (int j = 0; j < 8; ++j) {
        const int c = chalf * 128 + j * 16 + ln;
        const float bias = lds_b2[c];
#pragma unroll
        for (int e = 0; e < 8; ++e) {
            const int m = m0 + strip * 16 + e + 8 * lh;
            __builtin_nontemporal_store((acc2[j][e] + bias) * pv[e],
                                        &out[((size_t)t * kB + m) * kC + c]);
        }
    }
}

// ---------------------------------------------------------------------------
extern "C" void kernel_launch(void* const* d_in, const int* in_sizes, int n_in,
                              void* d_out, int out_size, void* d_ws, size_t ws_size,
                              hipStream_t stream) {
    const float* x  = (const float*)d_in[0];   // [B,D]
    const float* Wr = (const float*)d_in[1];   // [D,T]
    const float* br = (const float*)d_in[2];   // [T]
    const float* W1 = (const float*)d_in[3];   // [T,D,H]
    const float* b1 = (const float*)d_in[4];   // [T,H]
    const float* W2 = (const float*)d_in[5];   // [T,H,C]
    const float* b2 = (const float*)d_in[6];   // [T,C]

    float* probs = (float*)d_out;                           // [B,T]
    float* preds = (float*)d_out + (size_t)kB * kT;         // [T,B,C]

    // workspace: bf16 copies (x row-major; weights N-major) — ~21 MB
    __bf16* xb  = (__bf16*)d_ws;                            // [B,D]
    __bf16* w1t = xb + (size_t)kB * kD;                     // [T,H,D]
    __bf16* w2t = w1t + (size_t)kT * kH * kD;               // [T,C,H]

    cvt_x_kernel<<<(kB * kD) / 1024, 256, 0, stream>>>(x, xb);

    dim3 g1(kD / 32, kH / 32, kT);
    transpose_cvt_kernel<<<g1, 256, 0, stream>>>(W1, w1t, kD, kH);
    dim3 g2(kH / 32, kC / 32, kT);
    transpose_cvt_kernel<<<g2, 256, 0, stream>>>(W2, w2t, kH, kC);

    router_kernel<<<kB / 16, 256, 0, stream>>>(x, Wr, br, probs);

    dim3 ge(kB / 64, kT);
    expert_kernel<<<ge, 256, 0, stream>>>(xb, w1t, w2t, b1, b2, probs, preds);
}
